// Router_58849641889869
// MI455X (gfx1250) — matrix operand order repaired and checked by reading, not compiled, not run on hardware
//
#include <hip/hip_runtime.h>
#include <math.h>

// ---------------------------------------------------------------------------
// Router: logits = h @ W^T  (T=32768, D=2048, E=64), bias/pad, softmax, top-2.
// MI455X plan: bf16 WMMA (v_wmma_f32_16x16x32_bf16), W staged to LDS as bf16
// (double buffered), h streamed from HBM (the 256 MB roofline term).
// f32->bf16 via native v_cvt_pk_bf16_f32; B fragments batch-loaded per k-step
// so ds_load latency overlaps WMMA execution (staggered s_wait_dscnt).
// ---------------------------------------------------------------------------

#define T_TOK   32768
#define D_DIM   2048
#define E_EXP   64
#define KC      128              // K-chunk staged per double-buffer slot
#define NCHUNK  (D_DIM / KC)     // 16
#define LDW_U32 68               // (KC + 8 pad) bf16 per row / 2 = uints per row
#define TOK_PER_WAVE  16
#define TOK_PER_BLOCK 128        // 8 waves * 16 tokens

typedef __attribute__((ext_vector_type(16))) __bf16 v16bf;
typedef __attribute__((ext_vector_type(2)))  __bf16 v2bf;
typedef __attribute__((ext_vector_type(8)))  float  v8f;

union ABf16 { v16bf v; unsigned int u[8]; };

__device__ __forceinline__ unsigned int pk2bf(float a, float b) {
#if __has_builtin(__builtin_amdgcn_cvt_pk_bf16_f32)
    union { v2bf v; unsigned int u; } cv;
    cv.v = __builtin_amdgcn_cvt_pk_bf16_f32(a, b);
    return cv.u;
#else
    union { v2bf v; unsigned int u; } cv;
    cv.v.x = (__bf16)a;                   // native cvt, RNE
    cv.v.y = (__bf16)b;
    return cv.u;
#endif
}

// merge two (value,index)-sorted top-2 pairs; ties -> lower index (jax top_k)
__device__ __forceinline__ void top2merge(float& v1, int& i1, float& v2, int& i2,
                                          float ov1, int oi1, float ov2, int oi2) {
    if (ov1 > v1 || (ov1 == v1 && oi1 < i1)) {
        float nv2; int ni2;
        if (v1 > ov2 || (v1 == ov2 && i1 < oi2)) { nv2 = v1; ni2 = i1; }
        else                                      { nv2 = ov2; ni2 = oi2; }
        v1 = ov1; i1 = oi1; v2 = nv2; i2 = ni2;
    } else if (ov1 > v2 || (ov1 == v2 && oi1 < i2)) {
        v2 = ov1; i2 = oi1;
    }
}

// Stage one K-chunk of W (64 x 128 fp32) into LDS as bf16, row-major with
// +16B row pad.  256 threads, 8 float4 loads each; coalesced on K.
__device__ __forceinline__ void stageW(const float* __restrict__ W,
                                       unsigned int* __restrict__ dst,
                                       int chunk, int tid) {
#pragma unroll
    for (int i = 0; i < 8; ++i) {
        int idx = tid + i * 256;                 // 0..2047 float4 slots
        int row = idx >> 5;                      // expert 0..63
        int k4  = idx & 31;                      // float4 within chunk
        const float4 f = *(const float4*)(W + (size_t)row * D_DIM
                                            + chunk * KC + k4 * 4);
        uint2 p; p.x = pk2bf(f.x, f.y); p.y = pk2bf(f.z, f.w);
        *(uint2*)&dst[row * LDW_U32 + k4 * 2] = p;
    }
}

__global__ void __launch_bounds__(256)
router_kernel(const float* __restrict__ h,
              const int*   __restrict__ is_valid,
              const float* __restrict__ W,
              float* __restrict__ outMask,
              float* __restrict__ outProb) {
    __shared__ unsigned int Wlds[2][E_EXP * LDW_U32];   // 2 x 17 KB bf16 slabs

    const int tid    = threadIdx.x;
    const int wave   = tid >> 5;
    const int lane   = tid & 31;
    const int laneLo = lane & 15;
    const int hiHalf = lane >> 4;

    const int tokBase = blockIdx.x * TOK_PER_BLOCK + wave * TOK_PER_WAVE;
    const float* __restrict__ hRow = h + (size_t)(tokBase + laneLo) * D_DIM;

    v8f acc[4];
#pragma unroll
    for (int t = 0; t < 4; ++t) acc[t] = (v8f){0.f,0.f,0.f,0.f,0.f,0.f,0.f,0.f};

    stageW(W, Wlds[0], 0, tid);

    for (int c = 0; c < NCHUNK; ++c) {
        __syncthreads();
        if (c + 1 < NCHUNK) stageW(W, Wlds[(c + 1) & 1], c + 1, tid);
        const unsigned int* __restrict__ wb = Wlds[c & 1];

#pragma unroll
        for (int s = 0; s < KC / 32; ++s) {          // 4 WMMA K-steps per chunk
            const int kb = c * KC + s * 32 + hiHalf * 8;
            // ---- batch all 8 B ds_loads first: staggered dscnt waits -------
            uint4 braw[8];
#pragma unroll
            for (int t = 0; t < 4; ++t) {            // 4 expert tiles of 16
                const int n   = laneLo + (t << 4);   // B: lane -> column N
                const int off = n * LDW_U32 + s * 16 + hiHalf * 4;
                braw[2 * t]     = *(const uint4*)&wb[off];
                braw[2 * t + 1] = *(const uint4*)&wb[off + 8];
            }
            // ---- A fragment (16x32 bf16): lane holds row=laneLo,
            //      K = kb..kb+7 (elems 0-7) and kb+16..kb+23 (elems 8-15) ----
            const float4 a0 = *(const float4*)(hRow + kb);
            const float4 a1 = *(const float4*)(hRow + kb + 4);
            const float4 a2 = *(const float4*)(hRow + kb + 16);
            const float4 a3 = *(const float4*)(hRow + kb + 20);
            ABf16 A;
            A.u[0] = pk2bf(a0.x, a0.y); A.u[1] = pk2bf(a0.z, a0.w);
            A.u[2] = pk2bf(a1.x, a1.y); A.u[3] = pk2bf(a1.z, a1.w);
            A.u[4] = pk2bf(a2.x, a2.y); A.u[5] = pk2bf(a2.z, a2.w);
            A.u[6] = pk2bf(a3.x, a3.y); A.u[7] = pk2bf(a3.z, a3.w);

            // ---- fire the 4 WMMAs --------------------------------------
#pragma unroll
            for (int t = 0; t < 4; ++t) {
                ABf16 B;
                B.u[0] = braw[2 * t].x; B.u[1] = braw[2 * t].y;
                B.u[2] = braw[2 * t].z; B.u[3] = braw[2 * t].w;
                B.u[4] = braw[2 * t + 1].x; B.u[5] = braw[2 * t + 1].y;
                B.u[6] = braw[2 * t + 1].z; B.u[7] = braw[2 * t + 1].w;
                acc[t] = __builtin_amdgcn_wmma_f32_16x16x32_bf16(
                    false, A.v, false, B.v, (short)0, acc[t], false, false);
            }
        }
    }

    // ---------------- epilogue: bias, pad, softmax, top-2, masks ------------
    // C layout: lane laneLo holds expert e = 16t+laneLo; VGPR r is token row
    // M = r + 8*hiHalf.  Each 16-lane half reduces its 8 rows independently.
#pragma unroll
    for (int r = 0; r < 8; ++r) {
        const int tok = tokBase + r + hiHalf * 8;
        const bool valid = is_valid[tok] != 0;

        float vv[4];
#pragma unroll
        for (int t = 0; t < 4; ++t) vv[t] = acc[t][r];
        if (laneLo == 15) vv[3] += 0.5f;                    // ID_BIAS on e==63
        if (!valid) {
#pragma unroll
            for (int t = 0; t < 4; ++t) {
                const bool isId = (laneLo == 15) && (t == 3);
                vv[t] = isId ? (vv[t] + 1e30f) : -1e30f;
            }
        }

        // local top-2 over this lane's 4 experts
        float v1 = vv[0]; int i1 = laneLo;
        float v2 = -INFINITY; int i2 = 0x7fffffff;
#pragma unroll
        for (int t = 1; t < 4; ++t) {
            const int e = laneLo + (t << 4);
            if (vv[t] > v1 || (vv[t] == v1 && e < i1)) {
                v2 = v1; i2 = i1; v1 = vv[t]; i1 = e;
            } else if (vv[t] > v2 || (vv[t] == v2 && e < i2)) {
                v2 = vv[t]; i2 = e;
            }
        }
        // butterfly across the 16-lane half
#pragma unroll
        for (int m = 1; m < 16; m <<= 1) {
            const float ov1 = __shfl_xor(v1, m, 32);
            const int   oi1 = __shfl_xor(i1, m, 32);
            const float ov2 = __shfl_xor(v2, m, 32);
            const int   oi2 = __shfl_xor(i2, m, 32);
            top2merge(v1, i1, v2, i2, ov1, oi1, ov2, oi2);
        }

        const float rowmax = v1;                            // top-1 == row max
        float p[4]; float s = 0.f;
#pragma unroll
        for (int t = 0; t < 4; ++t) { p[t] = __expf(vv[t] - rowmax); s += p[t]; }
#pragma unroll
        for (int m = 1; m < 16; m <<= 1) s += __shfl_xor(s, m, 32);
        const float invs = __fdividef(1.0f, s);

        const size_t rowOff = (size_t)tok * E_EXP;
#pragma unroll
        for (int t = 0; t < 4; ++t) {
            const int e = laneLo + (t << 4);
            const bool mk = valid ? (e == i1 || e == i2) : (e == E_EXP - 1);
            outMask[rowOff + e] = mk ? 1.0f : 0.0f;
            outProb[rowOff + e] = p[t] * invs;
        }
    }
}

extern "C" void kernel_launch(void* const* d_in, const int* in_sizes, int n_in,
                              void* d_out, int out_size, void* d_ws, size_t ws_size,
                              hipStream_t stream) {
    const float* h        = (const float*)d_in[0];
    const int*   is_valid = (const int*)d_in[1];
    const float* W        = (const float*)d_in[2];
    float* outMask = (float*)d_out;
    float* outProb = outMask + (size_t)T_TOK * E_EXP;

    dim3 grid(T_TOK / TOK_PER_BLOCK);   // 256 blocks
    dim3 block(256);                    // 8 wave32
    router_kernel<<<grid, block, 0, stream>>>(h, is_valid, W, outMask, outProb);
}
